// Decoder_71949292142783
// MI455X (gfx1250) — matrix operand order repaired and checked by reading, not compiled, not run on hardware
//
#include <hip/hip_runtime.h>
#include <hip/hip_fp16.h>

typedef __attribute__((ext_vector_type(16))) _Float16 v16h;
typedef __attribute__((ext_vector_type(8)))  _Float16 v8h;
typedef __attribute__((ext_vector_type(4)))  _Float16 v4h;
typedef __attribute__((ext_vector_type(8)))  float    v8f;

#define E_EDGES 400000
#define NN      50000
#define HID     128
#define LN_EPS  1e-5f

// ---- workspace layout (bytes) ----
// [0, 25.6MB)           seg_sum  f32 [NN][128]
// [25.6MB, 25.8MB)      cnt      f32 [NN]
// [25.8MB, +294912B)    swizzled f16 weights
#define SEG_FLOATS ((size_t)NN * HID)
#define CNT_FLOATS ((size_t)NN)
#define W_OFF_BYTES ((SEG_FLOATS + CNT_FLOATS) * 4)   // 25,800,000 (32B aligned)
// f16 sub-offsets (in halves)
#define OFF_EEW2 0
#define OFF_PEW1 16384
#define OFF_PEW2 65536
#define OFF_PNW1 81920
#define OFF_PNW2 114688
#define OFF_NOW1 131072
#define W_TOTAL_HALVES 147456

__device__ __forceinline__ float sigmf(float x) { return 1.0f / (1.0f + __expf(-x)); }
__device__ __forceinline__ float siluf(float x) { return x * sigmf(x); }

__device__ __forceinline__ void st4h(_Float16* p, float4 v) {
    v4h h; h[0] = (_Float16)v.x; h[1] = (_Float16)v.y; h[2] = (_Float16)v.z; h[3] = (_Float16)v.w;
    *(v4h*)p = h;
}

// A fragment (16x32 f16) for lane L: rows = L&15, two contiguous 8-half chunks at
// K = k0 + 8*(L>>4) and K = k0 + 16 + 8*(L>>4)   (ISA 16-bit A-matrix table)
__device__ __forceinline__ v16h load_a_frag(const _Float16* rowBase, int stride, int k0, int lane) {
    const int row = lane & 15, hh = lane >> 4;
    const _Float16* p = rowBase + row * stride + k0 + 8 * hh;
    v8h lo = *(const v8h*)p;
    v8h hi = *(const v8h*)(p + 16);
    return __builtin_shufflevector(lo, hi, 0,1,2,3,4,5,6,7,8,9,10,11,12,13,14,15);
}

// C += A(16xK from LDS) * B(Kx128 pre-swizzled f16 fragments in global)
template <int KT>
__device__ __forceinline__ void gemm_tiles(const _Float16* aLds, int stride,
                                           const _Float16* bw, int lane, v8f acc[8]) {
    for (int kt = 0; kt < KT; ++kt) {
        v16h a = load_a_frag(aLds, stride, kt * 32, lane);
        const v16h* bp = ((const v16h*)bw) + (size_t)(kt * 8) * 32 + lane;
#pragma unroll
        for (int t = 0; t < 8; ++t) {
            v16h b = bp[t * 32];
            acc[t] = __builtin_amdgcn_wmma_f32_16x16x32_f16(
                false, a, false, b, (short)0, acc[t], false, false);
        }
    }
}

// in-place LayerNorm across the 128 columns of a 16x128 accumulator set
__device__ __forceinline__ void layernorm_acc(v8f acc[8], const float* g, const float* be, int lane) {
    float s1[8], s2[8];
#pragma unroll
    for (int j = 0; j < 8; ++j) { s1[j] = 0.f; s2[j] = 0.f; }
#pragma unroll
    for (int t = 0; t < 8; ++t)
#pragma unroll
        for (int j = 0; j < 8; ++j) { float v = acc[t][j]; s1[j] += v; s2[j] += v * v; }
#pragma unroll
    for (int m = 1; m < 16; m <<= 1)
#pragma unroll
        for (int j = 0; j < 8; ++j) {
            s1[j] += __shfl_xor(s1[j], m, 32);
            s2[j] += __shfl_xor(s2[j], m, 32);
        }
    const int n = lane & 15;
#pragma unroll
    for (int j = 0; j < 8; ++j) {
        float mu = s1[j] * (1.0f / 128.0f);
        float var = s2[j] * (1.0f / 128.0f) - mu * mu;
        s1[j] = mu;
        s2[j] = rsqrtf(var + LN_EPS);
    }
#pragma unroll
    for (int t = 0; t < 8; ++t) {
        float gg = g[16 * t + n], bb = be[16 * t + n];
#pragma unroll
        for (int j = 0; j < 8; ++j) acc[t][j] = (acc[t][j] - s1[j]) * s2[j] * gg + bb;
    }
}

// ---------- kernel 0: zero seg/cnt + swizzle weights f32->f16 fragment layout ----------
__device__ __forceinline__ void swz(const float* src, _Float16* dst, int KT,
                                    long long tid, long long stride) {
    long long total = (long long)KT * 4096;   // 32 rows x 128 cols per k-tile
    for (long long i = tid; i < total; i += stride) {
        int e = (int)(i & 15);
        int L = (int)((i >> 4) & 31);
        int t = (int)((i >> 9) & 7);
        int kt = (int)(i >> 12);
        int k = kt * 32 + (e & 7) + 8 * (L >> 4) + 16 * (e >> 3);
        int c = t * 16 + (L & 15);
        dst[i] = (_Float16)src[(size_t)k * 128 + c];
    }
}

__global__ void k_prep(const float* eeW2, const float* peW1, const float* peW2,
                       const float* pnW1, const float* pnW2, const float* noW1,
                       float* ws) {
    long long tid = (long long)blockIdx.x * blockDim.x + threadIdx.x;
    long long stride = (long long)gridDim.x * blockDim.x;
    long long zn = (long long)(SEG_FLOATS + CNT_FLOATS);
    for (long long i = tid; i < zn; i += stride) ws[i] = 0.0f;
    _Float16* wd = (_Float16*)((char*)ws + W_OFF_BYTES);
    swz(eeW2, wd + OFF_EEW2, 4,  tid, stride);
    swz(peW1, wd + OFF_PEW1, 12, tid, stride);
    swz(peW2, wd + OFF_PEW2, 4,  tid, stride);
    swz(pnW1, wd + OFF_PNW1, 8,  tid, stride);
    swz(pnW2, wd + OFF_PNW2, 4,  tid, stride);
    swz(noW1, wd + OFF_NOW1, 4,  tid, stride);
}

// ---------- kernel 1: edge encoder: edge_lat = LN(silu(x@W1+b1)@W2+b2) ----------
__global__ void __launch_bounds__(256)
k_edge_enc(const float* __restrict__ ef, const float* __restrict__ W1,
           const float* __restrict__ b1, const _Float16* __restrict__ W2s,
           const float* __restrict__ b2, const float* __restrict__ g,
           const float* __restrict__ be, float* __restrict__ edge_lat) {
    __shared__ __align__(16) _Float16 smem[8 * 16 * 128];
    const int lane = threadIdx.x & 31, wave = threadIdx.x >> 5;
    const int e0 = (blockIdx.x * 8 + wave) * 16;
    _Float16* my = smem + wave * 16 * 128;

    // h = silu(x@W1 + b1): 3-term VALU, 4 cols/lane/row
    const int c0 = lane * 4;
    float4 w0 = *(const float4*)(W1 + 0 * 128 + c0);
    float4 w1 = *(const float4*)(W1 + 1 * 128 + c0);
    float4 w2 = *(const float4*)(W1 + 2 * 128 + c0);
    float4 bb = *(const float4*)(b1 + c0);
    for (int r = 0; r < 16; ++r) {
        const float* x = ef + (size_t)(e0 + r) * 3;
        float x0 = x[0], x1 = x[1], x2 = x[2];
        float4 h;
        h.x = siluf(x0 * w0.x + x1 * w1.x + x2 * w2.x + bb.x);
        h.y = siluf(x0 * w0.y + x1 * w1.y + x2 * w2.y + bb.y);
        h.z = siluf(x0 * w0.z + x1 * w1.z + x2 * w2.z + bb.z);
        h.w = siluf(x0 * w0.w + x1 * w1.w + x2 * w2.w + bb.w);
        st4h(my + r * 128 + c0, h);
    }
    v8f acc[8] = {};
    gemm_tiles<4>(my, 128, W2s, lane, acc);
    const int n = lane & 15, hh = lane >> 4;
#pragma unroll
    for (int t = 0; t < 8; ++t) {
        float bv = b2[16 * t + n];
#pragma unroll
        for (int j = 0; j < 8; ++j) acc[t][j] += bv;
    }
    layernorm_acc(acc, g, be, lane);
#pragma unroll
    for (int t = 0; t < 8; ++t)
#pragma unroll
        for (int j = 0; j < 8; ++j)
            edge_lat[(size_t)(e0 + j + 8 * hh) * 128 + 16 * t + n] = acc[t][j];
}

// ---------- kernel 2: edge processor + fused segment-sum atomics ----------
__global__ void __launch_bounds__(128)
k_edge_proc(const int* __restrict__ eidx, const float* __restrict__ nodeLat,
            const float* __restrict__ nodeFeat, const float* __restrict__ edge_lat,
            const _Float16* __restrict__ W1s, const float* __restrict__ b1,
            const _Float16* __restrict__ W2s, const float* __restrict__ b2,
            const float* __restrict__ g, const float* __restrict__ be,
            float* __restrict__ seg, float* __restrict__ cnt) {
    __shared__ __align__(16) _Float16 smem[4 * 16 * 384];
    const int lane = threadIdx.x & 31, wave = threadIdx.x >> 5;
    const int e0 = (blockIdx.x * 4 + wave) * 16;
    _Float16* my = smem + wave * 16 * 384;

    // stage e_in = [sender_f | receiver_f | edge_lat] as f16 (coalesced float4 loads)
    for (int r = 0; r < 16; ++r) {
        int e = e0 + r;
        int s  = eidx[2 * e];
        int rc = eidx[2 * e + 1];
        float4 a = ((const float4*)(nodeLat  + (size_t)s  * 128))[lane];
        float4 b = ((const float4*)(nodeFeat + (size_t)rc * 128))[lane];
        float4 c = ((const float4*)(edge_lat + (size_t)e  * 128))[lane];
        st4h(my + r * 384 +       lane * 4, a);
        st4h(my + r * 384 + 128 + lane * 4, b);
        st4h(my + r * 384 + 256 + lane * 4, c);
    }
    v8f acc[8] = {};
    gemm_tiles<12>(my, 384, W1s, lane, acc);      // 384 -> 128
    const int n = lane & 15, hh = lane >> 4;
#pragma unroll
    for (int t = 0; t < 8; ++t) {
        float bv = b1[16 * t + n];
#pragma unroll
        for (int j = 0; j < 8; ++j)
            my[(j + 8 * hh) * 384 + 16 * t + n] = (_Float16)siluf(acc[t][j] + bv);
    }
    v8f acc2[8] = {};
    gemm_tiles<4>(my, 384, W2s, lane, acc2);      // 128 -> 128
#pragma unroll
    for (int t = 0; t < 8; ++t) {
        float bv = b2[16 * t + n];
#pragma unroll
        for (int j = 0; j < 8; ++j) acc2[t][j] += bv;
    }
    layernorm_acc(acc2, g, be, lane);

    // fused segment-sum: scatter-add new_edge rows into seg[receiver]
#pragma unroll
    for (int j = 0; j < 8; ++j) {
        int row = j + 8 * hh;
        int rc = eidx[2 * (e0 + row) + 1];
        float* dst = seg + (size_t)rc * 128 + n;
#pragma unroll
        for (int t = 0; t < 8; ++t)
            unsafeAtomicAdd(dst + 16 * t, acc2[t][j]);
    }
    if (lane < 16) unsafeAtomicAdd(cnt + eidx[2 * (e0 + lane) + 1], 1.0f);
}

// ---------- kernel 3: node update + output head ----------
__global__ void __launch_bounds__(128)
k_node(const float* __restrict__ nodeFeat, const float* __restrict__ seg,
       const float* __restrict__ cnt,
       const _Float16* __restrict__ pnW1s, const float* __restrict__ pnb1,
       const _Float16* __restrict__ pnW2s, const float* __restrict__ pnb2,
       const float* __restrict__ png, const float* __restrict__ pnbe,
       const _Float16* __restrict__ noW1s, const float* __restrict__ nob1,
       const float* __restrict__ noW2, const float* __restrict__ nob2,
       float* __restrict__ outp) {
    __shared__ __align__(16) _Float16 smem[4 * 16 * 256];
    const int lane = threadIdx.x & 31, wave = threadIdx.x >> 5;
    const int tile = blockIdx.x * 4 + wave;
    if (tile >= NN / 16) return;                  // whole-wave exit; EXEC stays full
    const int n0 = tile * 16;
    _Float16* my = smem + wave * 16 * 256;

    for (int r = 0; r < 16; ++r) {
        int node = n0 + r;
        float inv = 1.0f / fmaxf(cnt[node], 1.0f);
        float4 a = ((const float4*)(nodeFeat + (size_t)node * 128))[lane];
        float4 b = ((const float4*)(seg      + (size_t)node * 128))[lane];
        b.x *= inv; b.y *= inv; b.z *= inv; b.w *= inv;
        st4h(my + r * 256 +       lane * 4, a);
        st4h(my + r * 256 + 128 + lane * 4, b);
    }
    const int n = lane & 15, hh = lane >> 4;

    v8f acc[8] = {};
    gemm_tiles<8>(my, 256, pnW1s, lane, acc);     // 256 -> 128
#pragma unroll
    for (int t = 0; t < 8; ++t) {
        float bv = pnb1[16 * t + n];
#pragma unroll
        for (int j = 0; j < 8; ++j)
            my[(j + 8 * hh) * 256 + 16 * t + n] = (_Float16)siluf(acc[t][j] + bv);
    }
    v8f acc2[8] = {};
    gemm_tiles<4>(my, 256, pnW2s, lane, acc2);    // 128 -> 128
#pragma unroll
    for (int t = 0; t < 8; ++t) {
        float bv = pnb2[16 * t + n];
#pragma unroll
        for (int j = 0; j < 8; ++j) acc2[t][j] += bv;
    }
    layernorm_acc(acc2, png, pnbe, lane);
#pragma unroll
    for (int t = 0; t < 8; ++t)
#pragma unroll
        for (int j = 0; j < 8; ++j)
            my[(j + 8 * hh) * 256 + 16 * t + n] = (_Float16)acc2[t][j];   // new_node

    v8f acc3[8] = {};
    gemm_tiles<4>(my, 256, noW1s, lane, acc3);    // 128 -> 128 (output hidden)
#pragma unroll
    for (int t = 0; t < 8; ++t) {
        float bv = nob1[16 * t + n];
#pragma unroll
        for (int j = 0; j < 8; ++j)
            my[(j + 8 * hh) * 256 + 16 * t + n] = (_Float16)sigmf(acc3[t][j] + bv);
    }
    // final 128 -> 3 (VALU dot), 48 (row, out) pairs over 32 lanes
    for (int p = lane; p < 48; p += 32) {
        int r = p / 3, o = p % 3;
        float s = nob2[o];
        for (int c = 0; c < 128; ++c)
            s += (float)my[r * 256 + c] * noW2[(size_t)c * 3 + o];
        outp[(size_t)(n0 + r) * 3 + o] = s;
    }
}

extern "C" void kernel_launch(void* const* d_in, const int* in_sizes, int n_in,
                              void* d_out, int out_size, void* d_ws, size_t ws_size,
                              hipStream_t stream) {
    const int*   eidx  = (const int*)  d_in[0];
    const float* ef    = (const float*)d_in[1];
    const float* nlat  = (const float*)d_in[2];
    const float* nfeat = (const float*)d_in[3];
    const float* eeW1 = (const float*)d_in[4],  *eeb1 = (const float*)d_in[5];
    const float* eeW2 = (const float*)d_in[6],  *eeb2 = (const float*)d_in[7];
    const float* eeg  = (const float*)d_in[8],  *eebe = (const float*)d_in[9];
    const float* peW1 = (const float*)d_in[10], *peb1 = (const float*)d_in[11];
    const float* peW2 = (const float*)d_in[12], *peb2 = (const float*)d_in[13];
    const float* peg  = (const float*)d_in[14], *pebe = (const float*)d_in[15];
    const float* pnW1 = (const float*)d_in[16], *pnb1 = (const float*)d_in[17];
    const float* pnW2 = (const float*)d_in[18], *pnb2 = (const float*)d_in[19];
    const float* png  = (const float*)d_in[20], *pnbe = (const float*)d_in[21];
    const float* noW1 = (const float*)d_in[22], *nob1 = (const float*)d_in[23];
    const float* noW2 = (const float*)d_in[24], *nob2 = (const float*)d_in[25];

    float* seg = (float*)d_ws;
    float* cnt = seg + SEG_FLOATS;
    _Float16* wb = (_Float16*)((char*)d_ws + W_OFF_BYTES);
    _Float16* eeW2s = wb + OFF_EEW2;
    _Float16* peW1s = wb + OFF_PEW1;
    _Float16* peW2s = wb + OFF_PEW2;
    _Float16* pnW1s = wb + OFF_PNW1;
    _Float16* pnW2s = wb + OFF_PNW2;
    _Float16* noW1s = wb + OFF_NOW1;

    float* edge_lat = (float*)d_out;                       // 400000*128
    float* out2 = edge_lat + (size_t)E_EDGES * HID;        // 50000*3

    k_prep<<<2048, 256, 0, stream>>>(eeW2, peW1, peW2, pnW1, pnW2, noW1, (float*)d_ws);
    k_edge_enc<<<E_EDGES / (16 * 8), 256, 0, stream>>>(ef, eeW1, eeb1, eeW2s, eeb2, eeg, eebe,
                                                       edge_lat);
    k_edge_proc<<<E_EDGES / (16 * 4), 128, 0, stream>>>(eidx, nlat, nfeat, edge_lat,
                                                        peW1s, peb1, peW2s, peb2, peg, pebe,
                                                        seg, cnt);
    k_node<<<(NN / 16 + 3) / 4, 128, 0, stream>>>(nfeat, seg, cnt,
                                                  pnW1s, pnb1, pnW2s, pnb2, png, pnbe,
                                                  noW1s, nob1, noW2, nob2, out2);
}
